// Model_67525475828497
// MI455X (gfx1250) — matrix operand order repaired and checked by reading, not compiled, not run on hardware
//
#include <hip/hip_runtime.h>
#include <math.h>

// ---------------------------------------------------------------------------
// DCRNN (diffusion-conv GRU) for MI455X / gfx1250.
// Core compute: bf16 WMMA GEMMs (v_wmma_f32_16x16x32_bf16) with f32 accum.
// A-tile staging uses the CDNA5 async global->LDS path (ASYNCcnt).
// ---------------------------------------------------------------------------

typedef __attribute__((ext_vector_type(16))) __bf16 v16bf;
typedef __attribute__((ext_vector_type(8)))  float  v8f;

constexpr int kB    = 64;
constexpr int kTin  = 12;
constexpr int kTout = 12;
constexpr int kN    = 1024;
constexpr int kDin  = 2;
constexpr int kH    = 64;

constexpr int BM = 128, BN = 128, BK = 32;
constexpr int LDSTR = 36;   // bf16 elems per LDS tile row (32 + 4 pad: stride 18 dwords, conflict-free)

union FragU { unsigned int u[8]; v16bf v; };

// Async copy of 8 bytes global -> LDS, tracked by ASYNCcnt (bypasses VGPRs).
__device__ __forceinline__ void async_copy_b64(unsigned lds_addr, const void* gaddr)
{
  asm volatile("global_load_async_to_lds_b64 %0, %1, off"
               :: "v"(lds_addr), "v"(gaddr)
               : "memory");
}

__device__ __forceinline__ void wait_asynccnt0()
{
  asm volatile("s_wait_asynccnt 0x0" ::: "memory");
}

// D = alpha*(A@B) [+ betaE*E]          -> bf16 out   (diffusion / Chebyshev)
// D = (A@B) + bias[col]                -> f32  out   (weight multiply)
// A: [M,K] bf16 row-major (optional chunked K: element(r,k) = A[(k/chunkK)*chunkStride + r*lda + k%chunkK])
// B: [K,NN] bf16 row-major.  K must be a multiple of 32; M a multiple of 128; NN guarded.
template<bool OUT_BF16>
__global__ __launch_bounds__(256)
void wmma_gemm(const __bf16* __restrict__ A, int lda,
               const __bf16* __restrict__ Bm, int ldb,
               void* __restrict__ Cout, int ldc,
               const __bf16* __restrict__ E, int lde,
               const float* __restrict__ bias,
               float alpha, float betaE,
               int NN, int K,
               int chunkK, long long chunkStride)
{
  __shared__ __bf16 sA[BM * LDSTR];   // [m][k]
  __shared__ __bf16 sB[BN * LDSTR];   // [n][k]  (transposed tile)
  const int tid  = threadIdx.x;
  const int lane = tid & 31;
  const int wid  = tid >> 5;
  const int l16  = lane & 15;
  const int half = lane >> 4;
  const long long m0 = (long long)blockIdx.x * BM;
  const int n0 = blockIdx.y * BN;
  const int wm = (wid & 3) * 32;    // wave M offset: 4 waves along M
  const int wn = (wid >> 2) * 64;   // wave N offset: 2 waves along N

  v8f acc[2][4] = {};

  for (int k0 = 0; k0 < K; k0 += BK) {
    const __bf16* Abase;
    if (chunkK > 0) {
      int ch = k0 / chunkK;                       // BK tile never straddles a chunk (chunkK % 32 == 0)
      Abase = A + (long long)ch * chunkStride + (k0 - ch * chunkK);
    } else {
      Abase = A + k0;
    }
    // ---- stage A tile (128 x 32 bf16) via async global->LDS (ASYNCcnt) ----
    // 1024 segments of 4 bf16 (8B), 4 per thread; 8B alignment holds everywhere.
    #pragma unroll
    for (int i = 0; i < 4; ++i) {
      int seg = tid + i * 256;           // 0..1023
      int r   = seg >> 3;                // row 0..127
      int cs  = (seg & 7) * 4;           // col 0,4,...,28
      const __bf16* gp = Abase + (m0 + r) * (long long)lda + cs;
      unsigned lds = (unsigned)(size_t)&sA[r * LDSTR + cs];
      async_copy_b64(lds, (const void*)gp);
      if (k0 + BK < K)
        __builtin_prefetch((const void*)(gp + BK), 0, 0);
    }
    // ---- stage B tile (32 x 128 bf16) transposed into sB[n][k] ----
    #pragma unroll
    for (int i = 0; i < 8; ++i) {
      int idx  = tid + i * 256;          // 2048 dwords
      int kk   = idx >> 6;               // 0..31
      int nn2  = (idx & 63) * 2;         // even local col
      int gcol = n0 + nn2;
      unsigned int val = 0u;
      if (gcol < NN)
        val = *(const unsigned int*)(Bm + (long long)(k0 + kk) * ldb + gcol);
      sB[nn2 * LDSTR + kk]       = ((const __bf16*)&val)[0];
      sB[(nn2 + 1) * LDSTR + kk] = ((const __bf16*)&val)[1];
    }
    wait_asynccnt0();       // own wave's async LDS writes done
    __syncthreads();        // all waves' tile writes visible

    // ---- build fragments per ISA 16-bit A(16x32)/B(32x16) layouts ----
    FragU af[2], bfr[4];
    #pragma unroll
    for (int mt = 0; mt < 2; ++mt) {
      int row = wm + mt * 16 + l16;
      #pragma unroll
      for (int vi = 0; vi < 8; ++vi) {
        int kk = (vi & 3) * 2 + (vi >> 2) * 16 + half * 8;
        af[mt].u[vi] = *(const unsigned int*)&sA[row * LDSTR + kk];
      }
    }
    #pragma unroll
    for (int nt = 0; nt < 4; ++nt) {
      int col = wn + nt * 16 + l16;
      #pragma unroll
      for (int vi = 0; vi < 8; ++vi) {
        int kk = (vi & 3) * 2 + (vi >> 2) * 16 + half * 8;
        bfr[nt].u[vi] = *(const unsigned int*)&sB[col * LDSTR + kk];
      }
    }
    #pragma unroll
    for (int mt = 0; mt < 2; ++mt)
      #pragma unroll
      for (int nt = 0; nt < 4; ++nt)
        acc[mt][nt] = __builtin_amdgcn_wmma_f32_16x16x32_bf16(
            false, af[mt].v, false, bfr[nt].v, (short)0, acc[mt][nt], false, false);
    __syncthreads();
  }

  // ---- epilogue: C/D layout lanes0-15 M=r, lanes16-31 M=8+r, N=lane%16 ----
  #pragma unroll
  for (int mt = 0; mt < 2; ++mt) {
    #pragma unroll
    for (int nt = 0; nt < 4; ++nt) {
      int gcol = n0 + wn + nt * 16 + l16;
      if (gcol >= NN) continue;
      #pragma unroll
      for (int r = 0; r < 8; ++r) {
        long long grow = m0 + wm + mt * 16 + half * 8 + r;
        float v = acc[mt][nt][r];
        if (OUT_BF16) {
          float res = alpha * v;
          if (E) res += betaE * (float)E[grow * (long long)lde + gcol];
          ((__bf16*)Cout)[grow * (long long)ldc + gcol] = (__bf16)res;
        } else {
          float res = v + (bias ? bias[gcol] : 0.0f);
          ((float*)Cout)[grow * (long long)ldc + gcol] = res;
        }
      }
    }
  }
}

// ------------------------- elementwise / setup kernels ----------------------

__global__ void supports_sums(const float* __restrict__ adj,
                              float* __restrict__ rfw, float* __restrict__ rbw)
{
  int n = blockIdx.x * blockDim.x + threadIdx.x;
  if (n >= kN) return;
  float sfw = 0.f, sbw = 0.f;
  for (int j = 0; j < kN; ++j) {
    sfw += adj[(long long)n * kN + j];
    sbw += adj[(long long)j * kN + n];
  }
  rfw[n] = 1.0f / sfw;
  rbw[n] = 1.0f / sbw;
}

__global__ void supports_build(const float* __restrict__ adj,
                               const float* __restrict__ rfw, const float* __restrict__ rbw,
                               __bf16* __restrict__ Pfw, __bf16* __restrict__ Pbw)
{
  long long idx = (long long)blockIdx.x * blockDim.x + threadIdx.x;
  if (idx >= (long long)kN * kN) return;
  int i = (int)(idx / kN), j = (int)(idx % kN);
  // P_fw = (adj / rowsum)^T ; P_bw = (adj^T / colsum)^T
  Pfw[idx] = (__bf16)(adj[(long long)j * kN + i] * rfw[j]);
  Pbw[idx] = (__bf16)(adj[idx] * rbw[j]);
  (void)i;
}

// W[(c*5+m), out] -> Wr[m][cpad][out], zero pad rows
__global__ void weight_rearrange(const float* __restrict__ W, __bf16* __restrict__ Wr,
                                 int C, int Cpad, int OUT, int total)
{
  int idx = blockIdx.x * blockDim.x + threadIdx.x;
  if (idx >= total) return;
  int o = idx % OUT;
  int t = idx / OUT;
  int c = t % Cpad;
  int m = t / Cpad;
  float v = (c < C) ? W[((long long)c * 5 + m) * OUT + o] : 0.0f;
  Wr[idx] = (__bf16)v;
}

// X0[n,b,c] = concat(x, (r*)h) padded with zeros to Cpad.  x from batch_x (bx!=null)
// or from a [N,B,Dx] state buffer.  zu!=null -> multiply h by r-gate (first half of zu).
__global__ void build_xcat(const float* __restrict__ bx, int t,
                           const float* __restrict__ xbuf, int Dx,
                           const float* __restrict__ h,
                           const float* __restrict__ zu,
                           __bf16* __restrict__ X0, int Cpad, long long total)
{
  long long idx = (long long)blockIdx.x * blockDim.x + threadIdx.x;
  if (idx >= total) return;
  int c = (int)(idx % Cpad);
  long long nb = idx / Cpad;
  float v = 0.0f;
  if (c < Dx) {
    if (bx) {
      int b = (int)(nb % kB);
      int n = (int)(nb / kB);
      v = bx[(((long long)b * kTin + t) * kN + n) * kDin + c];
    } else {
      v = xbuf[nb * Dx + c];
    }
  } else if (c < Dx + kH) {
    int j = c - Dx;
    float hv = h[nb * kH + j];
    if (zu) hv *= zu[nb * (2 * kH) + j];
    v = hv;
  }
  X0[idx] = (__bf16)v;
}

__global__ void sigmoid_inplace(float* __restrict__ z, long long total)
{
  long long i = (long long)blockIdx.x * blockDim.x + threadIdx.x;
  if (i >= total) return;
  z[i] = 1.0f / (1.0f + __expf(-z[i]));
}

__global__ void cell_update(float* __restrict__ h, const float* __restrict__ zu,
                            const float* __restrict__ cc, long long total)
{
  long long i = (long long)blockIdx.x * blockDim.x + threadIdx.x;
  if (i >= total) return;
  long long nb = i / kH;
  int j = (int)(i % kH);
  float u = zu[nb * (2 * kH) + kH + j];
  float c = tanhf(cc[i]);
  h[i] = u * h[i] + (1.0f - u) * c;
}

__global__ void proj_kernel(const float* __restrict__ h, const float* __restrict__ pw,
                            const float* __restrict__ pb, float* __restrict__ go,
                            float* __restrict__ out, int t)
{
  long long nb = (long long)blockIdx.x * blockDim.x + threadIdx.x;
  if (nb >= (long long)kN * kB) return;
  int b = (int)(nb % kB);
  int n = (int)(nb / kB);
  float s = pb[0];
  #pragma unroll
  for (int j = 0; j < kH; ++j) s += h[nb * kH + j] * pw[j];
  go[nb] = s;
  out[((long long)b * kTout + t) * kN + n] = s;
}

__global__ void fill_zero_f32(float* __restrict__ p, long long total)
{
  long long i = (long long)blockIdx.x * blockDim.x + threadIdx.x;
  if (i < total) p[i] = 0.0f;
}

// --------------------------------- launcher ---------------------------------

extern "C" void kernel_launch(void* const* d_in, const int* in_sizes, int n_in,
                              void* d_out, int out_size, void* d_ws, size_t ws_size,
                              hipStream_t stream)
{
  (void)in_sizes; (void)n_in; (void)out_size; (void)ws_size;

  const float* batch_x = (const float*)d_in[0];
  const float* adj     = (const float*)d_in[1];
  const float* proj_W  = (const float*)d_in[18];
  const float* proj_b  = (const float*)d_in[19];

  struct Cell { const float *Wg,*bg,*Wc,*bc; __bf16 *Wrg,*Wrc; int Din, C, Cpad; };
  Cell cells[4];
  const int din_of[4] = { kDin, kH, 1, kH };   // enc0, enc1, dec0, dec1
  for (int i = 0; i < 4; ++i) {
    int base = 2 + i * 4;                      // Wg, bg, Wc, bc per cell
    cells[i].Wg  = (const float*)d_in[base + 0];
    cells[i].bg  = (const float*)d_in[base + 1];
    cells[i].Wc  = (const float*)d_in[base + 2];
    cells[i].bc  = (const float*)d_in[base + 3];
    cells[i].Din = din_of[i];
    cells[i].C   = din_of[i] + kH;
    cells[i].Cpad = ((cells[i].C + 31) / 32) * 32;
  }

  // ---- workspace bump allocator (256B aligned chunks) ----
  char* ws = (char*)d_ws;
  size_t off = 0;
  auto walloc = [&](size_t bytes) -> void* {
    void* p = ws + off;
    off += (bytes + 255) & ~(size_t)255;
    return p;
  };
  __bf16* Pfw = (__bf16*)walloc((size_t)kN * kN * 2);
  __bf16* Pbw = (__bf16*)walloc((size_t)kN * kN * 2);
  float*  rfw = (float*)walloc((size_t)kN * 4);
  float*  rbw = (float*)walloc((size_t)kN * 4);
  for (int i = 0; i < 4; ++i) {
    cells[i].Wrg = (__bf16*)walloc((size_t)5 * cells[i].Cpad * (2 * kH) * 2);
    cells[i].Wrc = (__bf16*)walloc((size_t)5 * cells[i].Cpad * kH * 2);
  }
  const long long NB = (long long)kN * kB;       // 65536 rows (n,b)
  float*  h0   = (float*)walloc((size_t)NB * kH * 4);
  float*  h1   = (float*)walloc((size_t)NB * kH * 4);
  float*  go   = (float*)walloc((size_t)NB * 4);
  __bf16* Xall = (__bf16*)walloc((size_t)5 * NB * 128 * 2);   // 5 diffusion chunks, max Cpad
  float*  zu   = (float*)walloc((size_t)NB * 2 * kH * 4);
  float*  cc   = (float*)walloc((size_t)NB * kH * 4);

  auto blocks = [](long long total) { return dim3((unsigned)((total + 255) / 256)); };

  // ---- supports ----
  supports_sums<<<dim3(4), 256, 0, stream>>>(adj, rfw, rbw);
  supports_build<<<blocks((long long)kN * kN), 256, 0, stream>>>(adj, rfw, rbw, Pfw, Pbw);

  // ---- weight rearrangement (c*5+m -> m-major, zero-padded K) ----
  for (int i = 0; i < 4; ++i) {
    long long tg = (long long)5 * cells[i].Cpad * (2 * kH);
    weight_rearrange<<<blocks(tg), 256, 0, stream>>>(cells[i].Wg, cells[i].Wrg,
                                                     cells[i].C, cells[i].Cpad, 2 * kH, (int)tg);
    long long tc = (long long)5 * cells[i].Cpad * kH;
    weight_rearrange<<<blocks(tc), 256, 0, stream>>>(cells[i].Wc, cells[i].Wrc,
                                                     cells[i].C, cells[i].Cpad, kH, (int)tc);
  }

  fill_zero_f32<<<blocks(NB * kH), 256, 0, stream>>>(h0, NB * kH);
  fill_zero_f32<<<blocks(NB * kH), 256, 0, stream>>>(h1, NB * kH);

  auto gemm_bf = [&](const __bf16* A, int lda, const __bf16* Bmat, int ldb,
                     __bf16* C, int ldc, const __bf16* E, int lde,
                     float alpha, float betaE, int M, int NN, int K) {
    dim3 g(M / BM, (NN + BN - 1) / BN);
    wmma_gemm<true><<<g, 256, 0, stream>>>(A, lda, Bmat, ldb, (void*)C, ldc, E, lde,
                                           nullptr, alpha, betaE, NN, K, 0, 0);
  };
  auto gemm_f32 = [&](const __bf16* A, int lda, const __bf16* Bmat, int ldb,
                      float* C, int ldc, const float* bias,
                      int M, int NN, int K, int chunkK, long long chunkStride) {
    dim3 g(M / BM, (NN + BN - 1) / BN);
    wmma_gemm<false><<<g, 256, 0, stream>>>(A, lda, Bmat, ldb, (void*)C, ldc, nullptr, 0,
                                            bias, 1.0f, 0.0f, NN, K, chunkK, chunkStride);
  };

  // Dual-support Chebyshev diffusion: X1=S1@X0, X2=2*S1@X1-X0, X3=S2@X0, X4=2*S2@X3-X0
  auto diffusion = [&](__bf16* X, long long NBCp, int BCp) {
    __bf16* X0 = X;
    __bf16* X1 = X + NBCp;
    __bf16* X2 = X + 2 * NBCp;
    __bf16* X3 = X + 3 * NBCp;
    __bf16* X4 = X + 4 * NBCp;
    gemm_bf(Pfw, kN, X0, BCp, X1, BCp, nullptr, 0, 1.0f,  0.0f, kN, BCp, kN);
    gemm_bf(Pfw, kN, X1, BCp, X2, BCp, X0,      BCp, 2.0f, -1.0f, kN, BCp, kN);
    gemm_bf(Pbw, kN, X0, BCp, X3, BCp, nullptr, 0, 1.0f,  0.0f, kN, BCp, kN);
    gemm_bf(Pbw, kN, X3, BCp, X4, BCp, X0,      BCp, 2.0f, -1.0f, kN, BCp, kN);
  };

  auto run_cell = [&](const Cell& c, const float* bx, int t, const float* xin, float* h) {
    const int Cpad = c.Cpad;
    const int BCp  = kB * Cpad;
    const long long NBCp = (long long)kN * BCp;
    // gate pass: zu = sigmoid(gconv([x,h]) Wg + bg)
    build_xcat<<<blocks(NBCp), 256, 0, stream>>>(bx, t, xin, c.Din, h, nullptr, Xall, Cpad, NBCp);
    diffusion(Xall, NBCp, BCp);
    gemm_f32(Xall, Cpad, c.Wrg, 2 * kH, zu, 2 * kH, c.bg,
             (int)NB, 2 * kH, 5 * Cpad, Cpad, NBCp);
    sigmoid_inplace<<<blocks(NB * 2 * kH), 256, 0, stream>>>(zu, NB * 2 * kH);
    // candidate pass: cc = gconv([x, r*h]) Wc + bc ; h = u*h + (1-u)*tanh(cc)
    build_xcat<<<blocks(NBCp), 256, 0, stream>>>(bx, t, xin, c.Din, h, zu, Xall, Cpad, NBCp);
    diffusion(Xall, NBCp, BCp);
    gemm_f32(Xall, Cpad, c.Wrc, kH, cc, kH, c.bc,
             (int)NB, kH, 5 * Cpad, Cpad, NBCp);
    cell_update<<<blocks(NB * kH), 256, 0, stream>>>(h, zu, cc, NB * kH);
  };

  // ---- encoder ----
  for (int t = 0; t < kTin; ++t) {
    run_cell(cells[0], batch_x, t, nullptr, h0);
    run_cell(cells[1], nullptr, 0, h0, h1);
  }
  // ---- decoder ----
  fill_zero_f32<<<blocks(NB), 256, 0, stream>>>(go, NB);
  float* out = (float*)d_out;
  for (int t = 0; t < kTout; ++t) {
    run_cell(cells[2], nullptr, 0, go, h0);
    run_cell(cells[3], nullptr, 0, h0, h1);
    proj_kernel<<<blocks(NB), 256, 0, stream>>>(h1, proj_W, proj_b, go, out, t);
  }
}
